// FunLinearNetwork_12352325944137
// MI455X (gfx1250) — matrix operand order repaired and checked by reading, not compiled, not run on hardware
//
#include <hip/hip_runtime.h>
#include <hip/hip_bf16.h>

// ---------------------------------------------------------------------------
// LayerNorm(4096) -> Linear(4096x4096, W stored [out,in]) + bias -> ReLU
// M = 4*4096 = 16384 tokens, K = N = 4096.
//
// MI455X (gfx1250, wave32) plan:
//   1) ln_bf16_kernel : fp32 LayerNorm, emit h as bf16 (workspace)
//   2) cvt_w_kernel   : W fp32 -> bf16 (32MB; lives in 192MB L2 for the GEMM)
//   3) gemm_wmma      : y = relu(h @ W^T + b) via v_wmma_f32_16x16x32_bf16,
//                       f32 accumulation, fused bias+ReLU epilogue.
//
// GEMM is compute-bound (0.55 TFLOP vs ~30us of HBM traffic at 23.3 TB/s),
// so the design maximizes WMMA issue density: wave tile 64x64 = 16 WMMA per
// 16 global_load_b128 (1 load/WMMA), 32-bit voffset addressing so the
// backend emits GLOBAL (not FLAT) loads.
// ---------------------------------------------------------------------------

#define DIM 4096
#define NTOK 16384  // 4 * 4096 tokens

typedef __attribute__((ext_vector_type(16))) __bf16 v16bf;
typedef __attribute__((ext_vector_type(8)))  __bf16 v8bf;
typedef __attribute__((ext_vector_type(4)))  __bf16 v4bf;
typedef __attribute__((ext_vector_type(8)))  float  v8f;

// ---------------------------------------------------------------------------
// Kernel 1: LayerNorm per row (4096 elems), one 256-thread block per row.
// ---------------------------------------------------------------------------
__global__ __launch_bounds__(256) void ln_bf16_kernel(
    const float* __restrict__ x, const float* __restrict__ gamma,
    const float* __restrict__ beta, __bf16* __restrict__ h) {
  const int row = blockIdx.x;
  const int t = threadIdx.x;
  const float* __restrict__ xr = x + (size_t)row * DIM;

  float v[16];
#pragma unroll
  for (int c = 0; c < 4; ++c) {
    const float4 f = *(const float4*)(xr + c * 1024 + t * 4);
    v[c * 4 + 0] = f.x; v[c * 4 + 1] = f.y;
    v[c * 4 + 2] = f.z; v[c * 4 + 3] = f.w;
  }

  __shared__ float red[8];
  const int wave = t >> 5, lane = t & 31;

  // mean
  float s = 0.f;
#pragma unroll
  for (int i = 0; i < 16; ++i) s += v[i];
#pragma unroll
  for (int off = 16; off > 0; off >>= 1) s += __shfl_xor(s, off, 32);
  if (lane == 0) red[wave] = s;
  __syncthreads();
  float tot = 0.f;
#pragma unroll
  for (int w = 0; w < 8; ++w) tot += red[w];
  const float mean = tot * (1.0f / DIM);
  __syncthreads();  // WAR before reusing red[]

  // variance
  float ss = 0.f;
#pragma unroll
  for (int i = 0; i < 16; ++i) {
    const float d = v[i] - mean;
    ss += d * d;
  }
#pragma unroll
  for (int off = 16; off > 0; off >>= 1) ss += __shfl_xor(ss, off, 32);
  if (lane == 0) red[wave] = ss;
  __syncthreads();
  float tot2 = 0.f;
#pragma unroll
  for (int w = 0; w < 8; ++w) tot2 += red[w];
  const float rstd = __frsqrt_rn(tot2 * (1.0f / DIM) + 1e-5f);

  __bf16* __restrict__ hr = h + (size_t)row * DIM;
#pragma unroll
  for (int c = 0; c < 4; ++c) {
    const int col = c * 1024 + t * 4;
    const float4 g4 = *(const float4*)(gamma + col);
    const float4 b4 = *(const float4*)(beta + col);
    v4bf o;
    o[0] = (__bf16)((v[c * 4 + 0] - mean) * rstd * g4.x + b4.x);
    o[1] = (__bf16)((v[c * 4 + 1] - mean) * rstd * g4.y + b4.y);
    o[2] = (__bf16)((v[c * 4 + 2] - mean) * rstd * g4.z + b4.z);
    o[3] = (__bf16)((v[c * 4 + 3] - mean) * rstd * g4.w + b4.w);
    *(v4bf*)(hr + col) = o;
  }
}

// ---------------------------------------------------------------------------
// Kernel 2: W fp32 -> bf16 (16M elements, one float4 per thread)
// ---------------------------------------------------------------------------
__global__ __launch_bounds__(256) void cvt_w_kernel(
    const float* __restrict__ W, __bf16* __restrict__ Wb) {
  const size_t i = ((size_t)blockIdx.x * 256 + threadIdx.x) * 4;
  const float4 f = *(const float4*)(W + i);
  v4bf o;
  o[0] = (__bf16)f.x; o[1] = (__bf16)f.y;
  o[2] = (__bf16)f.z; o[3] = (__bf16)f.w;
  *(v4bf*)(Wb + i) = o;
}

// ---------------------------------------------------------------------------
// Kernel 3: y = relu(h @ W^T + b) via WMMA bf16.
//
// Block tile 128(M) x 256(N), 256 threads = 8 waves arranged 2(M) x 4(N);
// each wave computes 64x64 = 4x4 tiles of 16x16 (16 x v8f accumulators).
// K loop: 32 per step (v_wmma_f32_16x16x32_bf16), 16 WMMA / 16 b128 loads.
//
// Fragment layouts per CDNA5 ISA 7.12.2 (wave32):
//  A 16x32 bf16 : lane l -> row (l&15); lanes 0-15 hold K[0..7],K[16..23],
//                 lanes 16-31 hold K[8..15],K[24..31]  (two b128 loads/lane)
//  B 32x16 bf16 : lane l -> col (l&15) == W row e (contiguous in K);
//                 lanes 0-15 K[0..7],K[16..23], lanes 16-31 K[8..15],K[24..31] (two b128 loads)
//  C/D 16x16 f32: lane l -> N=(l&15); VGPR v -> M = v + 8*(l>>4)
//
// All addressing is uniform-base + divergent 32-bit element offset so the
// backend can emit GLOBAL_LOAD_B128 (GVS: saddr + voffset), not FLAT.
// ---------------------------------------------------------------------------
__global__ __launch_bounds__(256) void gemm_wmma_kernel(
    const __bf16* __restrict__ h, const __bf16* __restrict__ Wb,
    const float* __restrict__ bias, float* __restrict__ y) {
  const int lane = threadIdx.x & 31;
  const int wave = threadIdx.x >> 5;
  const int wm = wave & 1;        // 0..1
  const int wn = wave >> 1;       // 0..3
  const int m0 = blockIdx.y * 128 + wm * 64;
  const int n0 = blockIdx.x * 256 + wn * 64;
  const int lr = lane & 15;       // row (A) / col (B) within a 16-tile
  const int hi = lane >> 4;       // K-half selector

  // 32-bit element offsets (h: 2^26 elems, Wb: 2^24 elems -> fit easily)
  unsigned aoff[4];
#pragma unroll
  for (int mi = 0; mi < 4; ++mi)
    aoff[mi] = (unsigned)((m0 + mi * 16 + lr) * DIM + hi * 8);
  unsigned boff[4];
#pragma unroll
  for (int ni = 0; ni < 4; ++ni)
    boff[ni] = (unsigned)((n0 + ni * 16 + lr) * DIM + hi * 8);

  v8f acc[4][4] = {};

  for (int k = 0; k < DIM; k += 32) {
    v16bf a[4], b[4];
#pragma unroll
    for (int mi = 0; mi < 4; ++mi) {
      const v8bf lo = *(const v8bf*)(h + aoff[mi]);
      const v8bf hc = *(const v8bf*)(h + aoff[mi] + 16);
#pragma unroll
      for (int i = 0; i < 8; ++i) { a[mi][i] = lo[i]; a[mi][i + 8] = hc[i]; }
      aoff[mi] += 32;
    }
#pragma unroll
    for (int ni = 0; ni < 4; ++ni) {
      const v8bf lo = *(const v8bf*)(Wb + boff[ni]);
      const v8bf hc = *(const v8bf*)(Wb + boff[ni] + 16);
#pragma unroll
      for (int i = 0; i < 8; ++i) { b[ni][i] = lo[i]; b[ni][i + 8] = hc[i]; }
      boff[ni] += 32;
    }
#pragma unroll
    for (int mi = 0; mi < 4; ++mi)
#pragma unroll
      for (int ni = 0; ni < 4; ++ni)
        acc[mi][ni] = __builtin_amdgcn_wmma_f32_16x16x32_bf16(
            /*neg_a=*/false, a[mi], /*neg_b=*/false, b[ni],
            /*c_mod=*/(short)0, acc[mi][ni],
            /*reuse_a=*/false, /*reuse_b=*/false);
  }

  // Epilogue: bias + ReLU, scatter per C/D layout.
  float bv[4];
#pragma unroll
  for (int ni = 0; ni < 4; ++ni) bv[ni] = bias[n0 + ni * 16 + lr];

#pragma unroll
  for (int mi = 0; mi < 4; ++mi) {
#pragma unroll
    for (int ni = 0; ni < 4; ++ni) {
      const int col = n0 + ni * 16 + lr;
#pragma unroll
      for (int v = 0; v < 8; ++v) {
        const int row = m0 + mi * 16 + v + hi * 8;
        const float val = acc[mi][ni][v] + bv[ni];
        y[(size_t)row * DIM + col] = fmaxf(val, 0.0f);
      }
    }
  }
}

// ---------------------------------------------------------------------------
// Host launcher
// ---------------------------------------------------------------------------
extern "C" void kernel_launch(void* const* d_in, const int* in_sizes, int n_in,
                              void* d_out, int out_size, void* d_ws, size_t ws_size,
                              hipStream_t stream) {
  const float* x     = (const float*)d_in[0];  // [4,4096,4096]
  const float* W     = (const float*)d_in[1];  // [4096,4096] (out,in)
  const float* b     = (const float*)d_in[2];  // [4096]
  const float* gamma = (const float*)d_in[3];  // [4096]
  const float* beta  = (const float*)d_in[4];  // [4096]
  float* y = (float*)d_out;                    // [4,4096,4096]

  // Workspace: h bf16 (128MB) then Wb bf16 (32MB)
  __bf16* hB = (__bf16*)d_ws;
  __bf16* wB = hB + (size_t)NTOK * DIM;

  ln_bf16_kernel<<<NTOK, 256, 0, stream>>>(x, gamma, beta, hB);
  cvt_w_kernel<<<(DIM * (DIM / 4)) / 256, 256, 0, stream>>>(W, wB);

  dim3 grid(DIM / 256, NTOK / 128);  // (16 N-tiles, 128 M-tiles)
  gemm_wmma_kernel<<<grid, 256, 0, stream>>>(hB, wB, b, y);
}
